// C3Net_30623116820560
// MI455X (gfx1250) — matrix operand order repaired and checked by reading, not compiled, not run on hardware
//
#include <hip/hip_runtime.h>

#define DEVINL __device__ __forceinline__

#define GLOBAL_AS __attribute__((address_space(1)))
#define LDS_AS    __attribute__((address_space(3)))

#if defined(__HIP_DEVICE_COMPILE__) && defined(__has_builtin)
#if __has_builtin(__builtin_amdgcn_global_load_async_to_lds_b32) && \
    __has_builtin(__builtin_amdgcn_s_wait_asynccnt)
#define USE_ASYNC 1
#endif
#endif
#ifndef USE_ASYNC
#define USE_ASYNC 0
#endif

typedef __attribute__((ext_vector_type(16))) _Float16 v16h;
typedef __attribute__((ext_vector_type(8)))  _Float16 v8h;
typedef __attribute__((ext_vector_type(8)))  float    v8f;

// softplus(x) - log(2), numerically stable
DEVINL float sspf(float x) {
  return fmaxf(x, 0.0f) + log1pf(__expf(-fabsf(x))) - 0.6931471805599453f;
}

DEVINL v16h cat16(v8h lo, v8h hi) {
  v16h r;
#pragma unroll
  for (int i = 0; i < 8; ++i) { r[i] = lo[i]; r[i + 8] = hi[i]; }
  return r;
}

DEVINL v8f wmma_f32(v16h a, v16h b, v8f c) {
  return __builtin_amdgcn_wmma_f32_16x16x32_f16(false, a, false, b, (short)0, c,
                                                false, false);
}

// Load one 16x32 f16 A-tile (row-major, rowStride in halves) into WMMA A layout:
// lane l holds row l%16; VGPR0-3 = K (l/16)*8.., VGPR4-7 = K 16+(l/16)*8..
DEVINL v16h loadA16x32(const _Float16* base, int rowStride) {
  int lane = threadIdx.x & 31;
  const _Float16* p = base + (lane & 15) * rowStride + (lane >> 4) * 8;
  return cat16(*(const v8h*)p, *(const v8h*)(p + 16));
}

// Load one 32x16 f16 B-tile from transposed weights wT[N][Kpad] (row-major):
// lane l holds column l%16, VGPR0-3 = K kchunk*32+(l/16)*8.., VGPR4-7 = K kchunk*32+16+(l/16)*8..
DEVINL v16h loadB32x16(const _Float16* wT, int Kpad, int ntile, int kchunk) {
  int lane = threadIdx.x & 31;
  const _Float16* p =
      wT + ((lane & 15) + ntile * 16) * Kpad + kchunk * 32 + (lane >> 4) * 8;
  return cat16(*(const v8h*)p, *(const v8h*)(p + 16));
}

// One dense layer on a 16-row tile: (16 x 32*nk) @ (32*nk x 64) + bias [-> ssp],
// result written row-major f16 [16][64] to per-wave LDS buffer `out`.
DEVINL void stage16x64(const v16h* a, int nk, const _Float16* wT, int Kpad,
                       const float* bias, bool act, _Float16* out) {
  int lane = threadIdx.x & 31;
#pragma unroll
  for (int s = 0; s < 4; ++s) {
    v8f c = {};
    for (int k = 0; k < nk; ++k) c = wmma_f32(a[k], loadB32x16(wT, Kpad, s, k), c);
    int n = s * 16 + (lane & 15);
    float bv = bias ? bias[n] : 0.0f;
#pragma unroll
    for (int j = 0; j < 8; ++j) {
      float x = c[j] + bv;
      if (act) x = sspf(x);
      out[(j + 8 * (lane >> 4)) * 64 + n] = (_Float16)x;
    }
  }
}

// ---------------- prep kernels ----------------

// dst[n][k] = w[k][n] as f16, zero-padded to Kpad columns
__global__ void k_wT(const float* __restrict__ w, _Float16* __restrict__ dst,
                     int K, int N, int Kpad) {
  int i = blockIdx.x * blockDim.x + threadIdx.x;
  if (i >= N * Kpad) return;
  int n = i / Kpad, k = i - n * Kpad;
  dst[i] = (k < K) ? (_Float16)w[k * N + n] : (_Float16)0.0f;
}

__global__ void k_copy64(float* __restrict__ dst, const float* __restrict__ src) {
  dst[threadIdx.x] = src[threadIdx.x];
}

// Solvent path + y_k[b] = solv[b] @ in2f_k  (tiny, scalar math)
__global__ void k_solv(const float* __restrict__ props, const int* __restrict__ sid,
                       const float* __restrict__ logP, const float* __restrict__ pampa,
                       const float* __restrict__ w1, const float* __restrict__ b1,
                       const float* __restrict__ w2, const float* __restrict__ b2,
                       const float* __restrict__ in2f0, const float* __restrict__ in2f1,
                       const float* __restrict__ in2f2, float* __restrict__ Y) {
  __shared__ float sp[5], h[16], solv[64];
  int b = blockIdx.x, t = threadIdx.x;
  int id = sid[b];
  if (t < 5) sp[t] = (id == 103) ? logP[t] : (id == 104 ? pampa[t] : props[b * 5 + t]);
  __syncthreads();
  if (t < 16) {
    float s = b1[t];
    for (int i = 0; i < 5; ++i) s += sp[i] * w1[i * 16 + t];
    h[t] = sspf(s);
  }
  __syncthreads();
  {
    float s = b2[t];
    for (int j = 0; j < 16; ++j) s += h[j] * w2[j * 64 + t];
    solv[t] = s;
  }
  __syncthreads();
  const float* in2f[3] = {in2f0, in2f1, in2f2};
  for (int k = 0; k < 3; ++k) {
    float y = 0.0f;
    for (int j = 0; j < 64; ++j) y += solv[j] * in2f[k][j * 64 + t];
    Y[(k * 64 + b) * 64 + t] = y;
  }
}

// ---------------- main kernels ----------------

// atom0 = Encoder @ Wa + ba : (8192x16)@(16x64), WMMA (K padded 16->32)
__global__ void k_atom(const float* __restrict__ enc, const _Float16* __restrict__ wT,
                       const float* __restrict__ bias, float* __restrict__ atom) {
  __shared__ alignas(16) _Float16 gA[4][16 * 32];
  int wave = threadIdx.x >> 5, lane = threadIdx.x & 31;
  int tile = blockIdx.x * 4 + wave;  // 512 tiles total
  int r0 = tile * 16;
  int m = lane & 15, hh = lane >> 4;
#pragma unroll
  for (int j = 0; j < 8; ++j) {
    int col = hh * 8 + j;
    gA[wave][m * 32 + col] = (_Float16)enc[(r0 + m) * 16 + col];
    gA[wave][m * 32 + 16 + col] = (_Float16)0.0f;  // K padding
  }
  v16h a = loadA16x32(gA[wave], 32);
#pragma unroll
  for (int s = 0; s < 4; ++s) {
    v8f c = {};
    c = wmma_f32(a, loadB32x16(wT, 32, s, 0), c);
    int n = s * 16 + m;
    float bv = bias[n];
#pragma unroll
    for (int j = 0; j < 8; ++j)
      atom[(r0 + j + 8 * hh) * 64 + n] = c[j] + bv;
  }
}

// W_bond = mask * f2(ssp(f1(bond_dense(emb[type])))) fused; one 16-edge tile / wave
__global__ void k_bondW(const int* __restrict__ type, const int* __restrict__ maskB,
                        const float* __restrict__ emb,
                        const _Float16* __restrict__ bondWT, const float* __restrict__ bond_b,
                        const _Float16* __restrict__ f1T, const float* __restrict__ f1b,
                        const _Float16* __restrict__ f2T, const float* __restrict__ f2b,
                        _Float16* __restrict__ Wbond) {
  __shared__ alignas(16) _Float16 gA[4][16 * 32];
  __shared__ alignas(16) _Float16 stg[4][16 * 64];
  __shared__ float msk[4][16];
  int wave = threadIdx.x >> 5, lane = threadIdx.x & 31;
  int tile = blockIdx.x * 4 + wave;  // 8192 tiles total
  int e0 = tile * 16;
  int m = lane & 15, hh = lane >> 4;
  int ty = type[e0 + m];
#pragma unroll
  for (int j = 0; j < 8; ++j) {
    int col = hh * 8 + j;
    gA[wave][m * 32 + col] = (_Float16)emb[ty * 16 + col];
    gA[wave][m * 32 + 16 + col] = (_Float16)0.0f;
  }
  if (hh == 0) msk[wave][m] = (float)maskB[e0 + m];
  v16h a[2];
  a[0] = loadA16x32(gA[wave], 32);
  stage16x64(a, 1, bondWT, 32, bond_b, false, stg[wave]);  // bond
  a[0] = loadA16x32(stg[wave], 64);
  a[1] = loadA16x32(stg[wave] + 32, 64);
  stage16x64(a, 2, f1T, 64, f1b, true, stg[wave]);         // h = ssp(f1)
  a[0] = loadA16x32(stg[wave], 64);
  a[1] = loadA16x32(stg[wave] + 32, 64);
  float mk[8];
#pragma unroll
  for (int j = 0; j < 8; ++j) mk[j] = msk[wave][j + 8 * hh];
#pragma unroll
  for (int s = 0; s < 4; ++s) {
    v8f c = {};
    c = wmma_f32(a[0], loadB32x16(f2T, 64, s, 0), c);
    c = wmma_f32(a[1], loadB32x16(f2T, 64, s, 1), c);
    int n = s * 16 + m;
    float bv = f2b[n];
#pragma unroll
    for (int j = 0; j < 8; ++j)
      Wbond[(e0 + j + 8 * hh) * 64 + n] = (_Float16)((c[j] + bv) * mk[j]);
  }
}

// One bond interaction iteration: block = one molecule, 8 waves x 16 atoms
__global__ void k_bond_iter(float* __restrict__ atom, const int* __restrict__ nbr,
                            const _Float16* __restrict__ Wbond,
                            const _Float16* __restrict__ in2fT,
                            const _Float16* __restrict__ f2outT, const float* __restrict__ f2out_b,
                            const _Float16* __restrict__ denseT, const float* __restrict__ dense_b) {
  __shared__ alignas(16) _Float16 atomA[128 * 64];
  __shared__ alignas(16) _Float16 ys[128 * 64];
  __shared__ alignas(16) _Float16 stg[8][16 * 64];
  int b = blockIdx.x;
  int wave = threadIdx.x >> 5, lane = threadIdx.x & 31;
  int m = lane & 15, hh = lane >> 4;
  const int base = b * 128 * 64;
  for (int i = threadIdx.x; i < 128 * 64; i += 256)
    atomA[i] = (_Float16)atom[base + i];
  __syncthreads();
  {  // y = atom @ in2f (no bias), wave w -> rows w*16..w*16+15
    const _Float16* Ab = atomA + wave * 16 * 64;
    v16h a0 = loadA16x32(Ab, 64), a1 = loadA16x32(Ab + 32, 64);
#pragma unroll
    for (int s = 0; s < 4; ++s) {
      v8f c = {};
      c = wmma_f32(a0, loadB32x16(in2fT, 64, s, 0), c);
      c = wmma_f32(a1, loadB32x16(in2fT, 64, s, 1), c);
      int n = s * 16 + m;
#pragma unroll
      for (int j = 0; j < 8; ++j)
        ys[(wave * 16 + j + 8 * hh) * 64 + n] = (_Float16)c[j];
    }
  }
  __syncthreads();
  // gather-weighted sum over 16 neighbors; lane covers 2 features
  for (int aa = 0; aa < 16; ++aa) {
    int a = wave * 16 + aa;
    int ebase = (b * 128 + a) * 16;
    int f = lane * 2;
    // speculative prefetch of the next atom's W rows (dropped if OOB)
    __builtin_prefetch((const void*)(Wbond + (ebase + 16) * 64), 0, 1);
    float acc0 = 0.0f, acc1 = 0.0f;
    for (int nn = 0; nn < 16; ++nn) {
      int idx = nbr[ebase + nn];
      const _Float16* wp = Wbond + (ebase + nn) * 64 + f;
      acc0 += (float)wp[0] * (float)ys[idx * 64 + f];
      acc1 += (float)wp[1] * (float)ys[idx * 64 + f + 1];
    }
    stg[wave][aa * 64 + f]     = (_Float16)acc0;
    stg[wave][aa * 64 + f + 1] = (_Float16)acc1;
  }
  v16h a[2];
  a[0] = loadA16x32(stg[wave], 64);
  a[1] = loadA16x32(stg[wave] + 32, 64);
  stage16x64(a, 2, f2outT, 64, f2out_b, true, stg[wave]);
  a[0] = loadA16x32(stg[wave], 64);
  a[1] = loadA16x32(stg[wave] + 32, 64);
#pragma unroll
  for (int s = 0; s < 4; ++s) {
    v8f c = {};
    c = wmma_f32(a[0], loadB32x16(denseT, 64, s, 0), c);
    c = wmma_f32(a[1], loadB32x16(denseT, 64, s, 1), c);
    int n = s * 16 + m;
    float bv = dense_b[n];
#pragma unroll
    for (int j = 0; j < 8; ++j) {
      int row = wave * 16 + j + 8 * hh;
      atom[base + row * 64 + n] += c[j] + bv;  // atom += v
    }
  }
}

// S_k[b,a,f] = sum_n mask * (f2(ssp(f1(g(r))))+b2)[f]; one (b,a) per wave.
// dist/mask tiles are double-buffered through LDS with async global->LDS loads.
__global__ void k_nonbondS(const float* __restrict__ dist, const int* __restrict__ maskNB,
                           const _Float16* __restrict__ aiF1T, const _Float16* __restrict__ aiF2T,
                           const float* __restrict__ biasAI, float* __restrict__ S) {
  __shared__ alignas(16) _Float16 gA[8][16 * 32];
  __shared__ alignas(16) _Float16 stg[8][16 * 64];
#if USE_ASYNC
  __shared__ alignas(16) unsigned dbuf[8][2][32];  // [wave][buf][0:15 dist,16:31 mask]
#endif
  int wave = threadIdx.x >> 5, lane = threadIdx.x & 31;
  int wid = blockIdx.x * 8 + wave;  // b*128 + a, 8192 total
  int m = lane & 15, hh = lane >> 4;
  const int ebase = wid * 128;
  float acc[3][4] = {};
#if USE_ASYNC
  {  // prologue: async-prefetch tile 0 (ASYNCcnt-tracked)
    const void* g = (lane < 16) ? (const void*)(dist + ebase + lane)
                                : (const void*)(maskNB + ebase + (lane - 16));
    __builtin_amdgcn_global_load_async_to_lds_b32(
        (GLOBAL_AS int*)(unsigned long long)g,
        (LDS_AS int*)(unsigned)(unsigned long long)(const void*)&dbuf[wave][0][lane],
        0, 0);
  }
#endif
  for (int nt = 0; nt < 8; ++nt) {
    int e0 = ebase + nt * 16;
    float r;
    float mk[8];
#if USE_ASYNC
    __builtin_amdgcn_s_wait_asynccnt(0);
    {
      const volatile unsigned* db = &dbuf[wave][nt & 1][0];
      r = __uint_as_float(db[m]);
#pragma unroll
      for (int j = 0; j < 8; ++j) mk[j] = (float)(int)db[16 + j + 8 * hh];
    }
    if (nt < 7) {  // async-prefetch the next tile while this one computes
      int e1 = e0 + 16;
      const void* g = (lane < 16) ? (const void*)(dist + e1 + lane)
                                  : (const void*)(maskNB + e1 + (lane - 16));
      __builtin_amdgcn_global_load_async_to_lds_b32(
          (GLOBAL_AS int*)(unsigned long long)g,
          (LDS_AS int*)(unsigned)(unsigned long long)(const void*)&dbuf[wave][(nt + 1) & 1][lane],
          0, 0);
    }
#else
    r = dist[e0 + m];
#pragma unroll
    for (int j = 0; j < 8; ++j) mk[j] = (float)maskNB[e0 + j + 8 * hh];
#endif
#pragma unroll
    for (int j = 0; j < 8; ++j) {  // gaussian smearing, NG=16 offsets on [0,1]
      int col = hh * 8 + j;
      float d = r - (float)col * (1.0f / 15.0f);
      gA[wave][m * 32 + col] = (_Float16)__expf(-112.5f * d * d);
      gA[wave][m * 32 + 16 + col] = (_Float16)0.0f;
    }
    v16h ag[2];
    ag[0] = loadA16x32(gA[wave], 32);
#pragma unroll
    for (int k = 0; k < 3; ++k) {
      stage16x64(ag, 1, aiF1T + k * 64 * 32, 32, biasAI + (k * 4 + 0) * 64, true, stg[wave]);
      v16h h[2];
      h[0] = loadA16x32(stg[wave], 64);
      h[1] = loadA16x32(stg[wave] + 32, 64);
      const _Float16* w2 = aiF2T + k * 64 * 64;
      const float* b2 = biasAI + (k * 4 + 1) * 64;
#pragma unroll
      for (int s = 0; s < 4; ++s) {
        v8f c = {};
        c = wmma_f32(h[0], loadB32x16(w2, 64, s, 0), c);
        c = wmma_f32(h[1], loadB32x16(w2, 64, s, 1), c);
        float bv = b2[s * 16 + m];
        float ssum = 0.0f;
#pragma unroll
        for (int j = 0; j < 8; ++j) ssum += (c[j] + bv) * mk[j];  // mask + reduce edges
        ssum += __shfl_xor(ssum, 16, 32);                         // cross-half rows
        acc[k][s] += ssum;
      }
    }
  }
  if (hh == 0) {
#pragma unroll
    for (int k = 0; k < 3; ++k)
#pragma unroll
      for (int s = 0; s < 4; ++s)
        S[(k * 8192 + wid) * 64 + s * 16 + m] = acc[k][s];
  }
}

// out = atom * sum_k dense_k(ssp(f2out_k(S_k * y_k)));  one 16-atom tile per wave
__global__ void k_final(const float* __restrict__ atom, const float* __restrict__ S,
                        const float* __restrict__ Y,
                        const _Float16* __restrict__ aiF2outT, const _Float16* __restrict__ aiDenseT,
                        const float* __restrict__ biasAI, float* __restrict__ out) {
  __shared__ alignas(16) _Float16 stg[8][16 * 64];
  int wave = threadIdx.x >> 5, lane = threadIdx.x & 31;
  int wid = blockIdx.x * 8 + wave;  // 512 tiles
  int b = wid >> 3, at = wid & 7;
  int rowbase = b * 128 + at * 16;
  int m = lane & 15, hh = lane >> 4;
  v8f vsum[4] = {};
  for (int k = 0; k < 3; ++k) {
    for (int i = lane; i < 16 * 64; i += 32) {
      int rr = i >> 6, f = i & 63;
      stg[wave][i] =
          (_Float16)(S[(k * 8192 + rowbase + rr) * 64 + f] * Y[(k * 64 + b) * 64 + f]);
    }
    v16h a[2];
    a[0] = loadA16x32(stg[wave], 64);
    a[1] = loadA16x32(stg[wave] + 32, 64);
    stage16x64(a, 2, aiF2outT + k * 64 * 64, 64, biasAI + (k * 4 + 2) * 64, true, stg[wave]);
    a[0] = loadA16x32(stg[wave], 64);
    a[1] = loadA16x32(stg[wave] + 32, 64);
    const _Float16* wd = aiDenseT + k * 64 * 64;
    const float* bd = biasAI + (k * 4 + 3) * 64;
#pragma unroll
    for (int s = 0; s < 4; ++s) {
      v8f c = {};
      c = wmma_f32(a[0], loadB32x16(wd, 64, s, 0), c);
      c = wmma_f32(a[1], loadB32x16(wd, 64, s, 1), c);
      float bv = bd[s * 16 + m];
#pragma unroll
      for (int j = 0; j < 8; ++j) vsum[s][j] += c[j] + bv;
    }
  }
#pragma unroll
  for (int s = 0; s < 4; ++s) {
    int n = s * 16 + m;
#pragma unroll
    for (int j = 0; j < 8; ++j) {
      int row = rowbase + j + 8 * hh;
      out[row * 64 + n] = atom[row * 64 + n] * vsum[s][j];
    }
  }
}

// ---------------- host ----------------

extern "C" void kernel_launch(void* const* d_in, const int* in_sizes, int n_in,
                              void* d_out, int out_size, void* d_ws, size_t ws_size,
                              hipStream_t stream) {
  (void)in_sizes; (void)n_in; (void)out_size; (void)ws_size;
  // top-level inputs (insertion order)
  const float* Encoder  = (const float*)d_in[0];
  const int*   nbrBond  = (const int*)d_in[1];
  const int*   maskBond = (const int*)d_in[2];
  const int*   typeBond = (const int*)d_in[3];
  const int*   maskNB   = (const int*)d_in[4];
  const float* distNB   = (const float*)d_in[5];
  const float* solvProp = (const float*)d_in[6];
  const int*   solvID   = (const int*)d_in[7];
  // params pytree leaves (dict keys sorted): PAMPA, ai1..ai3{dense(b,w),f1(b,w),
  // f2(b,w),f2out(b,w),in2f(w)}, atom_dense, bond_dense, bond_int, embedding,
  // logP_prop, solv1, solv2
  const float* PAMPA = (const float*)d_in[8];
  auto aip = [&](int k, int off) { return (const float*)d_in[9 + 9 * k + off]; };
  // off: 0=dense.b 1=dense.w 2=f1.b 3=f1.w 4=f2.b 5=f2.w 6=f2out.b 7=f2out.w 8=in2f.w
  const float* atom_b     = (const float*)d_in[36];
  const float* atom_w     = (const float*)d_in[37];
  const float* bond_b     = (const float*)d_in[38];
  const float* bond_w     = (const float*)d_in[39];
  const float* bi_dense_b = (const float*)d_in[40];
  const float* bi_dense_w = (const float*)d_in[41];
  const float* bi_f1_b    = (const float*)d_in[42];
  const float* bi_f1_w    = (const float*)d_in[43];
  const float* bi_f2_b    = (const float*)d_in[44];
  const float* bi_f2_w    = (const float*)d_in[45];
  const float* bi_f2out_b = (const float*)d_in[46];
  const float* bi_f2out_w = (const float*)d_in[47];
  const float* bi_in2f_w  = (const float*)d_in[48];
  const float* emb        = (const float*)d_in[49];
  const float* logP       = (const float*)d_in[50];
  const float* s1b        = (const float*)d_in[51];
  const float* s1w        = (const float*)d_in[52];
  const float* s2b        = (const float*)d_in[53];
  const float* s2w        = (const float*)d_in[54];

  char* ws = (char*)d_ws;
  size_t off = 0;
  auto carve = [&](size_t bytes) {
    off = (off + 255) & ~(size_t)255;
    char* p = ws + off;
    off += bytes;
    return p;
  };
  _Float16* atomWT   = (_Float16*)carve(64 * 32 * 2);
  _Float16* bondWT   = (_Float16*)carve(64 * 32 * 2);
  _Float16* biF1T    = (_Float16*)carve(64 * 64 * 2);
  _Float16* biF2T    = (_Float16*)carve(64 * 64 * 2);
  _Float16* biIn2fT  = (_Float16*)carve(64 * 64 * 2);
  _Float16* biF2outT = (_Float16*)carve(64 * 64 * 2);
  _Float16* biDenseT = (_Float16*)carve(64 * 64 * 2);
  _Float16* aiF1T    = (_Float16*)carve(3 * 64 * 32 * 2);
  _Float16* aiF2T    = (_Float16*)carve(3 * 64 * 64 * 2);
  _Float16* aiF2outT = (_Float16*)carve(3 * 64 * 64 * 2);
  _Float16* aiDenseT = (_Float16*)carve(3 * 64 * 64 * 2);
  _Float16* Wbond    = (_Float16*)carve((size_t)131072 * 64 * 2);
  float* biasAI = (float*)carve(3 * 4 * 64 * 4);
  float* Yk     = (float*)carve(3 * 64 * 64 * 4);
  float* atomB  = (float*)carve((size_t)8192 * 64 * 4);
  float* Sbuf   = (float*)carve((size_t)3 * 8192 * 64 * 4);

  auto wT = [&](const float* w, _Float16* dst, int K, int N, int Kpad) {
    int n = N * Kpad;
    k_wT<<<(n + 255) / 256, 256, 0, stream>>>(w, dst, K, N, Kpad);
  };
  wT(atom_w, atomWT, 16, 64, 32);
  wT(bond_w, bondWT, 16, 64, 32);
  wT(bi_f1_w, biF1T, 64, 64, 64);
  wT(bi_f2_w, biF2T, 64, 64, 64);
  wT(bi_in2f_w, biIn2fT, 64, 64, 64);
  wT(bi_f2out_w, biF2outT, 64, 64, 64);
  wT(bi_dense_w, biDenseT, 64, 64, 64);
  for (int k = 0; k < 3; ++k) {
    wT(aip(k, 3), aiF1T + k * 64 * 32, 16, 64, 32);
    wT(aip(k, 5), aiF2T + k * 64 * 64, 64, 64, 64);
    wT(aip(k, 7), aiF2outT + k * 64 * 64, 64, 64, 64);
    wT(aip(k, 1), aiDenseT + k * 64 * 64, 64, 64, 64);
    k_copy64<<<1, 64, 0, stream>>>(biasAI + (k * 4 + 0) * 64, aip(k, 2));  // f1.b
    k_copy64<<<1, 64, 0, stream>>>(biasAI + (k * 4 + 1) * 64, aip(k, 4));  // f2.b
    k_copy64<<<1, 64, 0, stream>>>(biasAI + (k * 4 + 2) * 64, aip(k, 6));  // f2out.b
    k_copy64<<<1, 64, 0, stream>>>(biasAI + (k * 4 + 3) * 64, aip(k, 0));  // dense.b
  }
  k_solv<<<64, 64, 0, stream>>>(solvProp, solvID, logP, PAMPA, s1w, s1b, s2w, s2b,
                                aip(0, 8), aip(1, 8), aip(2, 8), Yk);

  k_atom<<<128, 128, 0, stream>>>(Encoder, atomWT, atom_b, atomB);
  k_bondW<<<2048, 128, 0, stream>>>(typeBond, maskBond, emb, bondWT, bond_b, biF1T,
                                    bi_f1_b, biF2T, bi_f2_b, Wbond);
  for (int it = 0; it < 3; ++it)
    k_bond_iter<<<64, 256, 0, stream>>>(atomB, nbrBond, Wbond, biIn2fT, biF2outT,
                                        bi_f2out_b, biDenseT, bi_dense_b);
  k_nonbondS<<<1024, 256, 0, stream>>>(distNB, maskNB, aiF1T, aiF2T, biasAI, Sbuf);
  k_final<<<64, 256, 0, stream>>>(atomB, Sbuf, Yk, aiF2outT, aiDenseT, biasAI,
                                  (float*)d_out);
}
